// FlowGNN_76630806495926
// MI455X (gfx1250) — compile-verified, matrix operand order checked by reading, not run on hardware
//
#include <hip/hip_runtime.h>
#include <stdint.h>

typedef __attribute__((ext_vector_type(16))) __bf16 v16bf;
typedef __attribute__((ext_vector_type(8)))  __bf16 v8bf;
typedef __attribute__((ext_vector_type(8)))  float  v8f;

#define HID 128
#define NF 64
#define EF 64
#define FCOUT 16

// ---------------------------------------------------------------------------
// WMMA helpers (bf16 A/B, f32 accum, 16x16x32, wave32)
// A fragment layout (ISA 7.12.2, 16-bit A 16x32): lane L holds row m=L%16;
// halves 0..7 = K[hi8 .. hi8+7], halves 8..15 = K[16+hi8 .. 16+hi8+7],
// hi8 = (L>=16)*8.  Two aligned 16B LDS vector loads per fragment.
// ---------------------------------------------------------------------------
__device__ inline v16bf load_a_frag(const __bf16* row, int lane) {
    const int hi8 = (lane & 16) >> 1;   // 0 or 8
    v8bf lo = *(const v8bf*)(row + hi8);        // 16B aligned by construction
    v8bf hi = *(const v8bf*)(row + 16 + hi8);   // 16B aligned by construction
    return __builtin_shufflevector(lo, hi, 0, 1, 2, 3, 4, 5, 6, 7,
                                           8, 9, 10, 11, 12, 13, 14, 15);
}

__device__ inline v8f wmma_bf16(v16bf a, v16bf b, v8f c) {
    return __builtin_amdgcn_wmma_f32_16x16x32_bf16(false, a, false, b,
                                                   (short)0, c, false, false);
}

// ---------------------------------------------------------------------------
// Weight repack: fp32 [K x Nout] row-major (with up to 3 zero-padded concat
// segments) -> bf16 fragment layout [ktile][ntile][lane][16 halfs] so a GEMM
// lane loads 32 contiguous bytes per (ktile,ntile).
// ---------------------------------------------------------------------------
__global__ void k_repack(const float* __restrict__ W, __bf16* __restrict__ dst,
                         int Kpad, int Nout,
                         int s0len, int s0pad, int s1len, int s1pad, int s2len)
{
    int idx = blockIdx.x * blockDim.x + threadIdx.x;
    int total = Kpad * Nout;
    if (idx >= total) return;
    int k = idx / Nout, n = idx - k * Nout;
    int r = -1;
    if (k < s0pad)               { if (k < s0len) r = k; }
    else if (k < s0pad + s1pad)  { int kk = k - s0pad;         if (kk < s1len) r = s0len + kk; }
    else                         { int kk = k - s0pad - s1pad; if (kk < s2len) r = s0len + s1len + kk; }
    float v = (r >= 0) ? W[(size_t)r * Nout + n] : 0.0f;

    int kt = k >> 5, kl = k & 31;
    int nt = n >> 4, nl = n & 15;
    int jhi   = (kl >> 4) & 1;
    int rem   = kl & 15;
    int lhi   = (rem >> 3) & 1;
    int rem2  = rem & 7;
    int j     = jhi * 4 + (rem2 >> 1);
    int h     = rem2 & 1;
    int lanei = nl + lhi * 16;
    int NT    = Nout >> 4;
    size_t off = (((size_t)(kt * NT + nt) * 32 + lanei) * 16 + (j * 2 + h));
    dst[off] = (__bf16)v;
}

// ---------------------------------------------------------------------------
// Edge MLP: 32 edges per block (two 16-row M tiles share each B fragment).
// m = [h_in[src], h_in[dst], e] (zero padded to Kpad) -> relu(@W1+b1) [128]
// -> @W2+b2 [64] = new_e.  Writes new_e in place and atomic-adds aggr[dst].
// ---------------------------------------------------------------------------
__global__ __launch_bounds__(256)
void k_edge_mlp(const __bf16* __restrict__ hin, int ninp,
                const float* __restrict__ ein, int eindim,
                const int* __restrict__ src, const int* __restrict__ dst,
                const __bf16* __restrict__ w1p, const float* __restrict__ b1,
                const __bf16* __restrict__ w2p, const float* __restrict__ b2,
                float* __restrict__ eout, float* __restrict__ aggr,
                int E, int Kpad)
{
    __shared__ __align__(16) __bf16 sA[32 * 256];
    __shared__ __align__(16) __bf16 sH[32 * HID];
    __shared__ int sSrc[32], sDst[32];

    const int tid = threadIdx.x, lane = tid & 31, wv = tid >> 5;
    const int e0 = blockIdx.x * 32;

    if (tid < 32) {
        int e = e0 + tid; if (e >= E) e = E - 1;
        sSrc[tid] = src[e]; sDst[tid] = dst[e];
    }
    __syncthreads();

    const int twoN = 2 * ninp;
    for (int idx = tid; idx < 32 * Kpad; idx += 256) {
        int row = idx / Kpad, k = idx - row * Kpad;
        __bf16 v = (__bf16)0.0f;
        if (k < ninp)      v = hin[(size_t)sSrc[row] * ninp + k];
        else if (k < twoN) v = hin[(size_t)sDst[row] * ninp + (k - ninp)];
        else {
            int ke = k - twoN;
            if (ke < eindim) {
                int e = e0 + row; if (e >= E) e = E - 1;
                v = (__bf16)ein[(size_t)e * eindim + ke];
            }
        }
        sA[row * Kpad + k] = v;
    }
    __syncthreads();

    // GEMM1: [32 x Kpad] @ [Kpad x 128]; wave w owns columns w*16..w*16+15
    const int m = lane & 15;
    const int mh = (lane & 16) >> 1;
    const int Ktiles = Kpad >> 5;
    v8f acc0 = {}, acc1 = {};
    for (int kt = 0; kt < Ktiles; ++kt) {
        v16bf b  = *(const v16bf*)(w1p + (((size_t)kt * 8 + wv) * 32 + lane) * 16);
        v16bf a0 = load_a_frag(sA + m * Kpad + kt * 32, lane);
        v16bf a1 = load_a_frag(sA + (16 + m) * Kpad + kt * 32, lane);
        acc0 = wmma_bf16(a0, b, acc0);
        acc1 = wmma_bf16(a1, b, acc1);
    }
    {
        const int n = wv * 16 + m;
        const float bias = b1[n];
#pragma unroll
        for (int r = 0; r < 8; ++r) {
            float v0 = acc0[r] + bias; v0 = v0 > 0.0f ? v0 : 0.0f;
            float v1 = acc1[r] + bias; v1 = v1 > 0.0f ? v1 : 0.0f;
            sH[(r + mh) * HID + n]      = (__bf16)v0;
            sH[(16 + r + mh) * HID + n] = (__bf16)v1;
        }
    }
    __syncthreads();

    // GEMM2: [32 x 128] @ [128 x 64]; waves 0..3
    if (wv < 4) {
        v8f c0 = {}, c1 = {};
#pragma unroll
        for (int kt = 0; kt < 4; ++kt) {
            v16bf b  = *(const v16bf*)(w2p + (((size_t)kt * 4 + wv) * 32 + lane) * 16);
            v16bf a0 = load_a_frag(sH + m * HID + kt * 32, lane);
            v16bf a1 = load_a_frag(sH + (16 + m) * HID + kt * 32, lane);
            c0 = wmma_bf16(a0, b, c0);
            c1 = wmma_bf16(a1, b, c1);
        }
        const int n = wv * 16 + m;
        const float bias = b2[n];
#pragma unroll
        for (int r = 0; r < 8; ++r) {
            int row0 = r + mh, row1 = 16 + r + mh;
            int ea = e0 + row0, eb = e0 + row1;
            if (ea < E) {
                float v = c0[r] + bias;
                eout[(size_t)ea * EF + n] = v;
                atomicAdd(&aggr[(size_t)sSrc ? (size_t)sDst[row0] * EF + n : 0], v);
            }
            if (eb < E) {
                float v = c1[r] + bias;
                eout[(size_t)eb * EF + n] = v;
                atomicAdd(&aggr[(size_t)sDst[row1] * EF + n], v);
            }
        }
    }
}

// ---------------------------------------------------------------------------
// Node MLP: 32 nodes per block, [h_in, aggr] -> 128 -> 64
// ---------------------------------------------------------------------------
__global__ __launch_bounds__(256)
void k_node_mlp(const __bf16* __restrict__ hin, int ninp,
                const float* __restrict__ aggr,
                const __bf16* __restrict__ w1p, const float* __restrict__ b1,
                const __bf16* __restrict__ w2p, const float* __restrict__ b2,
                float* __restrict__ xout, int N, int Kpad)
{
    __shared__ __align__(16) __bf16 sA[32 * 160];
    __shared__ __align__(16) __bf16 sH[32 * HID];
    const int tid = threadIdx.x, lane = tid & 31, wv = tid >> 5;
    const int n0 = blockIdx.x * 32;

    for (int idx = tid; idx < 32 * Kpad; idx += 256) {
        int row = idx / Kpad, k = idx - row * Kpad;
        int node = n0 + row; if (node >= N) node = N - 1;
        __bf16 v;
        if (k < ninp) v = hin[(size_t)node * ninp + k];
        else          v = (__bf16)aggr[(size_t)node * NF + (k - ninp)];
        sA[row * Kpad + k] = v;
    }
    __syncthreads();

    const int m = lane & 15;
    const int mh = (lane & 16) >> 1;
    const int Ktiles = Kpad >> 5;
    v8f acc0 = {}, acc1 = {};
    for (int kt = 0; kt < Ktiles; ++kt) {
        v16bf b  = *(const v16bf*)(w1p + (((size_t)kt * 8 + wv) * 32 + lane) * 16);
        v16bf a0 = load_a_frag(sA + m * Kpad + kt * 32, lane);
        v16bf a1 = load_a_frag(sA + (16 + m) * Kpad + kt * 32, lane);
        acc0 = wmma_bf16(a0, b, acc0);
        acc1 = wmma_bf16(a1, b, acc1);
    }
    {
        const int n = wv * 16 + m;
        const float bias = b1[n];
#pragma unroll
        for (int r = 0; r < 8; ++r) {
            float v0 = acc0[r] + bias; v0 = v0 > 0.0f ? v0 : 0.0f;
            float v1 = acc1[r] + bias; v1 = v1 > 0.0f ? v1 : 0.0f;
            sH[(r + mh) * HID + n]      = (__bf16)v0;
            sH[(16 + r + mh) * HID + n] = (__bf16)v1;
        }
    }
    __syncthreads();

    if (wv < 4) {
        v8f c0 = {}, c1 = {};
#pragma unroll
        for (int kt = 0; kt < 4; ++kt) {
            v16bf b  = *(const v16bf*)(w2p + (((size_t)kt * 4 + wv) * 32 + lane) * 16);
            v16bf a0 = load_a_frag(sH + m * HID + kt * 32, lane);
            v16bf a1 = load_a_frag(sH + (16 + m) * HID + kt * 32, lane);
            c0 = wmma_bf16(a0, b, c0);
            c1 = wmma_bf16(a1, b, c1);
        }
        const int n = wv * 16 + m;
        const float bias = b2[n];
#pragma unroll
        for (int r = 0; r < 8; ++r) {
            int na = n0 + r + mh, nb = n0 + 16 + r + mh;
            if (na < N) xout[(size_t)na * NF + n] = c0[r] + bias;
            if (nb < N) xout[(size_t)nb * NF + n] = c1[r] + bias;
        }
    }
}

// ---------------------------------------------------------------------------
// Elementwise / scatter helper kernels
// ---------------------------------------------------------------------------
__global__ void k_zero(float* __restrict__ p, int n) {
    int i = blockIdx.x * blockDim.x + threadIdx.x;
    if (i < n) p[i] = 0.0f;
}

__global__ void k_deg(float* __restrict__ deg, const int* __restrict__ dst, int E) {
    int i = blockIdx.x * blockDim.x + threadIdx.x;
    if (i < E) atomicAdd(&deg[dst[i]], 1.0f);
}

__global__ void k_cnt(float* __restrict__ cnt, const int* __restrict__ batch, int N) {
    int i = blockIdx.x * blockDim.x + threadIdx.x;
    if (i < N) atomicAdd(&cnt[batch[i]], 1.0f);
}

__global__ void k_gp_acc(float* __restrict__ gp, const float* __restrict__ x,
                         const int* __restrict__ batch, int N) {
    int i = blockIdx.x * blockDim.x + threadIdx.x;
    if (i >= N * NF) return;
    int n = i >> 6, c = i & 63;
    atomicAdd(&gp[batch[n] * NF + c], x[i]);
}

__global__ void k_nb_acc(float* __restrict__ nbsum, const float* __restrict__ x,
                         const int* __restrict__ src, const int* __restrict__ dst, int E) {
    int i = blockIdx.x * blockDim.x + threadIdx.x;
    if (i >= E * NF) return;
    int e = i >> 6, c = i & 63;
    atomicAdd(&nbsum[(size_t)dst[e] * NF + c], x[(size_t)src[e] * NF + c]);
}

__global__ void k_node_smooth(float* __restrict__ xA, const float* __restrict__ xB,
                              const float* __restrict__ nbsum, const float* __restrict__ deg,
                              int N) {
    int i = blockIdx.x * blockDim.x + threadIdx.x;
    if (i >= N * NF) return;
    int n = i >> 6;
    float d = deg[n]; d = d > 1.0f ? d : 1.0f;
    xA[i] = 0.5f * (xB[i] + nbsum[i] / d);
}

__global__ void k_edge_smooth(float* __restrict__ e, const float* __restrict__ x,
                              const int* __restrict__ src, const int* __restrict__ dst, int E) {
    int i = blockIdx.x * blockDim.x + threadIdx.x;
    if (i >= E * NF) return;
    int ed = i >> 6, c = i & 63;
    e[i] = 0.5f * e[i] + 0.25f * (x[(size_t)src[ed] * NF + c] + x[(size_t)dst[ed] * NF + c]);
}

// h_in = [x (xdim), skip (2, = input x[:, :2]), optional fc_out[batch] (16)], padded to ninp
__global__ void k_prep_hin(__bf16* __restrict__ hin,
                           const float* __restrict__ xsrc, int xdim, int xstride,
                           const float* __restrict__ x_in,
                           const float* __restrict__ fcout,
                           const int* __restrict__ batch,
                           int ninp, int N)
{
    int idx = blockIdx.x * blockDim.x + threadIdx.x;
    if (idx >= N * ninp) return;
    int n = idx / ninp, k = idx - n * ninp;
    float v = 0.0f;
    if (k < xdim)               v = xsrc[(size_t)n * xstride + k];
    else if (k < xdim + 2)      v = x_in[(size_t)n * 6 + (k - xdim)];
    else if (fcout && k < xdim + 2 + FCOUT)
                                v = fcout[batch[n] * FCOUT + (k - xdim - 2)];
    hin[idx] = (__bf16)v;
}

__global__ void k_decode(float* __restrict__ out, const float* __restrict__ x,
                         const float* __restrict__ W, const float* __restrict__ b, int N) {
    int i = blockIdx.x * blockDim.x + threadIdx.x;
    if (i >= N * 4) return;
    int n = i >> 2, c = i & 3;
    float s = b[c];
#pragma unroll
    for (int k = 0; k < NF; ++k) s += x[(size_t)n * NF + k] * W[k * 4 + c];
    out[i] = s;
}

// ---------------------------------------------------------------------------
// Tiny per-batch (B=8) scalar kernels: conv encoder + fc blocks.
// ---------------------------------------------------------------------------
__global__ __launch_bounds__(256)
void k_encoder(const float* __restrict__ bc,
               const float* w1, const float* b1, const float* w2, const float* b2,
               const float* w3, const float* b3, const float* w4, const float* b4,
               float* __restrict__ h1, float* __restrict__ h2,
               float* __restrict__ h3, float* __restrict__ enc)
{
    const int tid = threadIdx.x;
    for (int idx = tid; idx < 8 * 10 * 122; idx += 256) {          // conv1 (1->10, s1)
        int b = idx / (10 * 122), r = idx % (10 * 122), o = r / 122, t = r % 122;
        float s = b1[o];
        for (int k = 0; k < 7; ++k) s += bc[b * 128 + t + k] * w1[o * 7 + k];
        h1[idx] = s;
    }
    __syncthreads();
    for (int idx = tid; idx < 8 * 20 * 58; idx += 256) {           // conv2 (10->20, s2)
        int b = idx / (20 * 58), r = idx % (20 * 58), o = r / 58, t = r % 58;
        float s = b2[o];
        for (int i = 0; i < 10; ++i)
            for (int k = 0; k < 7; ++k)
                s += h1[(b * 10 + i) * 122 + 2 * t + k] * w2[(o * 10 + i) * 7 + k];
        h2[idx] = s;
    }
    __syncthreads();
    for (int idx = tid; idx < 8 * 10 * 26; idx += 256) {           // conv3 (20->10, s2)
        int b = idx / (10 * 26), r = idx % (10 * 26), o = r / 26, t = r % 26;
        float s = b3[o];
        for (int i = 0; i < 20; ++i)
            for (int k = 0; k < 7; ++k)
                s += h2[(b * 20 + i) * 58 + 2 * t + k] * w3[(o * 20 + i) * 7 + k];
        h3[idx] = s;
    }
    __syncthreads();
    for (int idx = tid; idx < 8 * 20; idx += 256) {                // conv4 (10->1, s1)
        int b = idx / 20, t = idx % 20;
        float s = b4[0];
        for (int i = 0; i < 10; ++i)
            for (int k = 0; k < 7; ++k)
                s += h3[(b * 10 + i) * 26 + t + k] * w4[i * 7 + k];
        enc[idx] = s;
    }
}

__global__ __launch_bounds__(128)
void k_fc0(const float* __restrict__ enc,
           const float* W0, const float* b0, const float* W1, const float* b1,
           const float* W2, const float* b2, float* __restrict__ fcout)
{
    __shared__ float sin[20]; __shared__ float h0[128]; __shared__ float h1s[128];
    int tid = threadIdx.x;
    for (int b = 0; b < 8; ++b) {
        if (tid < 20) sin[tid] = enc[b * 20 + tid];
        __syncthreads();
        float s = b0[tid];
        for (int k = 0; k < 20; ++k) s += sin[k] * W0[k * 128 + tid];
        h0[tid] = s > 0.0f ? s : 0.0f;
        __syncthreads();
        s = b1[tid];
        for (int k = 0; k < 128; ++k) s += h0[k] * W1[k * 128 + tid];
        h1s[tid] = s > 0.0f ? s : 0.0f;
        __syncthreads();
        if (tid < 16) {
            s = b2[tid];
            for (int k = 0; k < 128; ++k) s += h1s[k] * W2[k * 16 + tid];
            fcout[b * 16 + tid] = s;
        }
        __syncthreads();
    }
}

__global__ __launch_bounds__(128)
void k_fc1(const float* __restrict__ bc, const float* __restrict__ fc0,
           const float* __restrict__ gp, const float* __restrict__ cnt,
           const float* W0, const float* b0, const float* W1, const float* b1,
           const float* W2, const float* b2, float* __restrict__ fcout)
{
    __shared__ float sin[208]; __shared__ float h0[128]; __shared__ float h1s[128];
    int tid = threadIdx.x;
    for (int b = 0; b < 8; ++b) {
        float c = cnt[b]; c = c > 1.0f ? c : 1.0f;
        for (int k = tid; k < 208; k += 128) {
            float v;
            if (k < 128)      v = bc[b * 128 + k];
            else if (k < 144) v = fc0[b * 16 + (k - 128)];
            else              v = gp[b * 64 + (k - 144)] / c;
            sin[k] = v;
        }
        __syncthreads();
        float s = b0[tid];
        for (int k = 0; k < 208; ++k) s += sin[k] * W0[k * 128 + tid];
        h0[tid] = s > 0.0f ? s : 0.0f;
        __syncthreads();
        s = b1[tid];
        for (int k = 0; k < 128; ++k) s += h0[k] * W1[k * 128 + tid];
        h1s[tid] = s > 0.0f ? s : 0.0f;
        __syncthreads();
        if (tid < 16) {
            s = b2[tid];
            for (int k = 0; k < 128; ++k) s += h1s[k] * W2[k * 16 + tid];
            fcout[b * 16 + tid] = s;
        }
        __syncthreads();
    }
}

// ---------------------------------------------------------------------------
// Host orchestration
// ---------------------------------------------------------------------------
extern "C" void kernel_launch(void* const* d_in, const int* in_sizes, int n_in,
                              void* d_out, int out_size, void* d_ws, size_t ws_size,
                              hipStream_t stream)
{
    const float* x_in  = (const float*)d_in[0];
    const int*   eidx  = (const int*)d_in[1];
    const float* eattr = (const float*)d_in[2];
    const float* bc    = (const float*)d_in[3];
    const int*   batch = (const int*)d_in[4];
    const int N = in_sizes[0] / 6;
    const int E = in_sizes[1] / 2;
    const int* src  = eidx;
    const int* dstp = eidx + E;

    int p = 5;
    const float* ew[4]; const float* ebv[4];   // enc conv weights/biases
    for (int i = 0; i < 4; ++i) { ew[i] = (const float*)d_in[p++]; ebv[i] = (const float*)d_in[p++]; }
    struct LayerP { const float *eW1, *eb1, *eW2, *eb2, *nW1, *nb1, *nW2, *nb2; } L[4];
    for (int i = 0; i < 4; ++i) {
        L[i].eW1 = (const float*)d_in[p++]; L[i].eb1 = (const float*)d_in[p++];
        L[i].eW2 = (const float*)d_in[p++]; L[i].eb2 = (const float*)d_in[p++];
        L[i].nW1 = (const float*)d_in[p++]; L[i].nb1 = (const float*)d_in[p++];
        L[i].nW2 = (const float*)d_in[p++]; L[i].nb2 = (const float*)d_in[p++];
    }
    struct FcP { const float *W0, *b0, *W1, *b1, *W2, *b2; } F[2];
    for (int i = 0; i < 2; ++i) {
        F[i].W0 = (const float*)d_in[p++]; F[i].b0 = (const float*)d_in[p++];
        F[i].W1 = (const float*)d_in[p++]; F[i].b1 = (const float*)d_in[p++];
        F[i].W2 = (const float*)d_in[p++]; F[i].b2 = (const float*)d_in[p++];
    }
    const float* decW = (const float*)d_in[p++];
    const float* decb = (const float*)d_in[p++];

    // workspace carve-out
    char* ws = (char*)d_ws;
    size_t off = 0;
    auto alloc = [&](size_t bytes) -> char* {
        char* r = ws + off;
        off = (off + bytes + 255) & ~(size_t)255;
        return r;
    };
    float*  eBuf  = (float*)alloc((size_t)E * EF * 4);
    float*  xA    = (float*)alloc((size_t)N * NF * 4);
    float*  xB    = (float*)alloc((size_t)N * NF * 4);
    float*  aggr  = (float*)alloc((size_t)N * EF * 4);
    float*  nbsum = (float*)alloc((size_t)N * NF * 4);
    __bf16* hin   = (__bf16*)alloc((size_t)N * 96 * 2);
    float*  deg   = (float*)alloc((size_t)N * 4);
    float*  cnt   = (float*)alloc(64 * 4);
    float*  h1    = (float*)alloc(8 * 10 * 122 * 4);
    float*  h2    = (float*)alloc(8 * 20 * 58 * 4);
    float*  h3    = (float*)alloc(8 * 10 * 26 * 4);
    float*  enc   = (float*)alloc(8 * 20 * 4);
    float*  fc0o  = (float*)alloc(8 * 16 * 4);
    float*  fc1o  = (float*)alloc(8 * 16 * 4);
    float*  gp    = (float*)alloc(8 * 64 * 4);

    const int nin_a [4] = {24, 66, 82, 66};
    const int ninp_a[4] = {32, 96, 96, 96};
    const int ein_a [4] = {3, 64, 64, 64};
    const int einp_a[4] = {32, 64, 64, 64};
    __bf16 *eW1p[4], *eW2p[4], *nW1p[4], *nW2p[4];
    for (int i = 0; i < 4; ++i) {
        int KpE = 2 * ninp_a[i] + einp_a[i];
        int KpN = ninp_a[i] + 64;
        eW1p[i] = (__bf16*)alloc((size_t)(KpE / 32) * 8 * 512 * 2);
        eW2p[i] = (__bf16*)alloc((size_t)4 * 4 * 512 * 2);
        nW1p[i] = (__bf16*)alloc((size_t)(KpN / 32) * 8 * 512 * 2);
        nW2p[i] = (__bf16*)alloc((size_t)4 * 4 * 512 * 2);
    }

    // --- encoder + fc0 + structural constants (deg, cnt) ---
    k_encoder<<<1, 256, 0, stream>>>(bc, ew[0], ebv[0], ew[1], ebv[1],
                                     ew[2], ebv[2], ew[3], ebv[3], h1, h2, h3, enc);
    k_fc0<<<1, 128, 0, stream>>>(enc, F[0].W0, F[0].b0, F[0].W1, F[0].b1, F[0].W2, F[0].b2, fc0o);
    k_zero<<<(N + 255) / 256, 256, 0, stream>>>(deg, N);
    k_zero<<<1, 64, 0, stream>>>(cnt, 64);
    k_deg<<<(E + 255) / 256, 256, 0, stream>>>(deg, dstp, E);
    k_cnt<<<(N + 255) / 256, 256, 0, stream>>>(cnt, batch, N);

    // --- repack all layer weights to WMMA B-fragment layout (bf16) ---
    for (int i = 0; i < 4; ++i) {
        int KpE = 2 * ninp_a[i] + einp_a[i];
        int KpN = ninp_a[i] + 64;
        k_repack<<<(KpE * 128 + 255) / 256, 256, 0, stream>>>(
            L[i].eW1, eW1p[i], KpE, 128,
            nin_a[i], ninp_a[i], nin_a[i], ninp_a[i], ein_a[i]);
        k_repack<<<(128 * 64 + 255) / 256, 256, 0, stream>>>(
            L[i].eW2, eW2p[i], 128, 64, 128, 128, 0, 0, 0);
        k_repack<<<(KpN * 128 + 255) / 256, 256, 0, stream>>>(
            L[i].nW1, nW1p[i], KpN, 128,
            nin_a[i], ninp_a[i], 64, 64, 0);
        k_repack<<<(128 * 64 + 255) / 256, 256, 0, stream>>>(
            L[i].nW2, nW2p[i], 128, 64, 128, 128, 0, 0, 0);
    }

    // --- 4 message-passing layers ---
    for (int i = 0; i < 4; ++i) {
        const int ninp = ninp_a[i];
        if (i == 2) {
            k_zero<<<(8 * 64 + 255) / 256, 256, 0, stream>>>(gp, 8 * 64);
            k_gp_acc<<<(N * NF + 255) / 256, 256, 0, stream>>>(gp, xA, batch, N);
            k_fc1<<<1, 128, 0, stream>>>(bc, fc0o, gp, cnt,
                                         F[1].W0, F[1].b0, F[1].W1, F[1].b1, F[1].W2, F[1].b2, fc1o);
        }
        const float* fcptr = (i == 0) ? fc0o : ((i == 2) ? fc1o : nullptr);
        const float* xs = (i == 0) ? x_in : xA;
        const int xdim = (i == 0) ? 6 : 64;
        k_prep_hin<<<(N * ninp + 255) / 256, 256, 0, stream>>>(
            hin, xs, xdim, xdim, x_in, fcptr, batch, ninp, N);

        k_zero<<<(N * EF + 255) / 256, 256, 0, stream>>>(aggr, N * EF);
        k_edge_mlp<<<(E + 31) / 32, 256, 0, stream>>>(
            hin, ninp, (i == 0) ? eattr : eBuf, ein_a[i],
            src, dstp, eW1p[i], L[i].eb1, eW2p[i], L[i].eb2,
            eBuf, aggr, E, 2 * ninp + einp_a[i]);

        k_node_mlp<<<(N + 31) / 32, 256, 0, stream>>>(
            hin, ninp, aggr, nW1p[i], L[i].nb1, nW2p[i], L[i].nb2, xB, N, ninp + 64);

        k_zero<<<(N * NF + 255) / 256, 256, 0, stream>>>(nbsum, N * NF);
        k_nb_acc<<<(E * NF + 255) / 256, 256, 0, stream>>>(nbsum, xB, src, dstp, E);
        k_node_smooth<<<(N * NF + 255) / 256, 256, 0, stream>>>(xA, xB, nbsum, deg, N);
        k_edge_smooth<<<(E * NF + 255) / 256, 256, 0, stream>>>(eBuf, xA, src, dstp, E);
    }

    // --- decode ---
    k_decode<<<(N * 4 + 255) / 256, 256, 0, stream>>>((float*)d_out, xA, decW, decb, N);
}
